// DiffusionLayer_89249420411227
// MI455X (gfx1250) — matrix operand order repaired and checked by reading, not compiled, not run on hardware
//
#include <hip/hip_runtime.h>
#include <hip/hip_bf16.h>

// ---------------- types for WMMA ----------------
typedef __attribute__((ext_vector_type(16))) _Float16 v16h;
typedef __attribute__((ext_vector_type(8)))  _Float16 v8h;
typedef __attribute__((ext_vector_type(8)))  float    v8f;

#define DIM 64
#define NU_ 200000
#define NI_ 100000
#define E_  1000000
#define WSTRIDE 72   // padded LDS row stride (halfs) -> 144B, conflict-free & 16B aligned

// ---------------- helpers ----------------
__device__ __forceinline__ unsigned encf(float f) {
    unsigned b = __float_as_uint(f);
    return (b & 0x80000000u) ? ~b : (b | 0x80000000u);
}
__device__ __forceinline__ float decf(unsigned u) {
    return __uint_as_float((u & 0x80000000u) ? (u & 0x7fffffffu) : ~u);
}
__device__ __forceinline__ float lrelu02(float x) { return x > 0.f ? x : 0.2f * x; }
__device__ __forceinline__ float lrelu001(float x) { return x > 0.f ? x : 0.01f * x; }

// ---------------- tiny utility kernels ----------------
__global__ void __launch_bounds__(256) k_f32_to_f16(const float* __restrict__ s,
                                                    _Float16* __restrict__ d, int n) {
    int i = blockIdx.x * 256 + threadIdx.x;
    if (i < n) d[i] = (_Float16)s[i];
}

__global__ void __launch_bounds__(256) k_zero_f32(float* __restrict__ p, int n) {
    int i = blockIdx.x * 256 + threadIdx.x;
    if (i < n) p[i] = 0.f;
}

__global__ void __launch_bounds__(256) k_init_mz(unsigned* __restrict__ m,
                                                 float* __restrict__ z, int n) {
    int i = blockIdx.x * 256 + threadIdx.x;
    if (i < n) { m[i] = 0u; z[i] = 0.f; }
}

// acc[i] += bias[i%64] (+ emb[i] if emb != nullptr)
__global__ void __launch_bounds__(256) k_add_bias_emb(float* __restrict__ acc,
                                                      const float* __restrict__ bias,
                                                      const float* __restrict__ emb, int n) {
    int i = blockIdx.x * 256 + threadIdx.x;
    if (i < n) {
        float v = acc[i] + bias[i & (DIM - 1)];
        if (emb) v += emb[i];
        acc[i] = v;
    }
}

// ---------------- WMMA GEMM: out[N,64] = A(f16)[N,64] @ W(f32)[64,64] + b ----------------
// W is transposed into LDS (Wt[n][k], padded stride) so each lane's B fragment is
// two contiguous 16B ds_load_b128 instead of 16 scalar ds_load_u16.
__global__ void __launch_bounds__(256)
k_gemm64_wmma(const _Float16* __restrict__ A, const float* __restrict__ W,
              const float* __restrict__ bias, float* __restrict__ out, int N) {
    __shared__ alignas(16) _Float16 Wl[DIM * WSTRIDE];
    int t = threadIdx.x;
    for (int i = t; i < DIM * DIM; i += 256) {
        int k = i >> 6;          // row of W  (K index)
        int n = i & (DIM - 1);   // col of W  (N index)
        Wl[n * WSTRIDE + k] = (_Float16)W[i];
    }
    __syncthreads();

    const int wave = t >> 5;
    const int lane = t & 31;
    const int hi   = (lane >> 4) & 1;     // 0: lanes 0-15, 1: lanes 16-31
    const int n16  = lane & 15;
    const int rowBase = blockIdx.x * 128 + wave * 16;

    int r  = rowBase + (lane & 15);
    int rc = r < N ? r : N - 1;           // clamp for safe loads; garbage rows never stored

    // A fragments (ISA 16-bit A 16x32 layout): lane<16 -> K 0-7 & 16-23, lane>=16 -> +8
    v16h afrag[2];
#pragma unroll
    for (int kb = 0; kb < 2; ++kb) {
        int base = kb * 32 + hi * 8;
        v8h lo = *(const v8h*)(A + rc * DIM + base);
        v8h hv = *(const v8h*)(A + rc * DIM + base + 16);
        v16h a;
#pragma unroll
        for (int i = 0; i < 8; ++i) { a[i] = lo[i]; a[i + 8] = hv[i]; }
        afrag[kb] = a;
    }

    v8f acc[4];
#pragma unroll
    for (int nt = 0; nt < 4; ++nt)
#pragma unroll
        for (int i = 0; i < 8; ++i) acc[nt][i] = 0.f;

#pragma unroll
    for (int nt = 0; nt < 4; ++nt) {
        int n = nt * 16 + n16;
        const _Float16* wrow = Wl + n * WSTRIDE;
#pragma unroll
        for (int kb = 0; kb < 2; ++kb) {
            int kbase = kb * 32 + hi * 8;    // B element e holds K = kbase + e (e<8), kbase + 16 + (e-8) (e>=8), col = n
            v8h b0 = *(const v8h*)(wrow + kbase);
            v8h b1 = *(const v8h*)(wrow + kbase + 16);
            v16h b;
#pragma unroll
            for (int i = 0; i < 8; ++i) { b[i] = b0[i]; b[i + 8] = b1[i]; }
            acc[nt] = __builtin_amdgcn_wmma_f32_16x16x32_f16(
                false, afrag[kb], false, b, (short)0, acc[nt], false, false);
        }
    }

    // Epilogue: wave-uniform fast path (no per-store exec churn) for full tiles.
    if (rowBase + 16 <= N) {
#pragma unroll
        for (int nt = 0; nt < 4; ++nt) {
            int n = nt * 16 + n16;
            float bv = bias[n];
#pragma unroll
            for (int v = 0; v < 8; ++v) {
                int row = rowBase + v + hi * 8;   // C/D layout: VGPR v -> M = v (+8 hi)
                out[row * DIM + n] = acc[nt][v] + bv;
            }
        }
    } else {
#pragma unroll
        for (int nt = 0; nt < 4; ++nt) {
            int n = nt * 16 + n16;
            float bv = bias[n];
#pragma unroll
            for (int v = 0; v < 8; ++v) {
                int row = rowBase + v + hi * 8;
                if (row < N) out[row * DIM + n] = acc[nt][v] + bv;
            }
        }
    }
}

// ---------------- edge passes (wave per edge) ----------------
__global__ void __launch_bounds__(256)
k_edge_pass1(const float* __restrict__ fs, const float* __restrict__ fd,
             const float* __restrict__ attn, const int* __restrict__ src,
             const int* __restrict__ dst, float* __restrict__ eval,
             unsigned* __restrict__ menc, int E) {
    int wave = threadIdx.x >> 5, lane = threadIdx.x & 31;
    int e = blockIdx.x * 8 + wave;
    if (e >= E) return;
    int s = src[e], d = dst[e];
    int k = lane * 2;
    float2 a  = *(const float2*)(fs + s * DIM + k);
    float2 b  = *(const float2*)(fd + d * DIM + k);
    float2 at = *(const float2*)(attn + k);
    float sum = at.x * lrelu02(a.x + b.x) + at.y * lrelu02(a.y + b.y);
#pragma unroll
    for (int off = 16; off; off >>= 1) sum += __shfl_xor(sum, off, 32);
    if (lane == 0) {
        eval[e] = sum;
        atomicMax(menc + d, encf(sum));
    }
}

__global__ void __launch_bounds__(256)
k_edge_pass2(float* __restrict__ eval, const int* __restrict__ dst,
             const unsigned* __restrict__ menc, float* __restrict__ z, int E) {
    int e = blockIdx.x * 256 + threadIdx.x;
    if (e >= E) return;
    int d = dst[e];
    float a = __expf(eval[e] - decf(menc[d]));
    eval[e] = a;
    atomicAdd(z + d, a);
}

__global__ void __launch_bounds__(256)
k_edge_pass3(const float* __restrict__ fs, const float* __restrict__ eval,
             const float* __restrict__ z, const int* __restrict__ src,
             const int* __restrict__ dst, float* __restrict__ acc, int E) {
    int wave = threadIdx.x >> 5, lane = threadIdx.x & 31;
    int e = blockIdx.x * 8 + wave;
    if (e >= E) return;
    int s = src[e], d = dst[e];
    float alpha = eval[e] / z[d];
    int k = lane * 2;
    float2 f = *(const float2*)(fs + s * DIM + k);
    atomicAdd(acc + d * DIM + k,     alpha * f.x);
    atomicAdd(acc + d * DIM + k + 1, alpha * f.y);
}

// ---------------- fused attention MLP ----------------
// small layout (floats): [0:128) v_inf, [128:256) v_int, [256] c_inf, [257] c_int,
//                        [260:264) stats {S1_inf, S2_inf, S1_int, S2_int}
__global__ void __launch_bounds__(256)
k_mlp_prep(const float* __restrict__ W1i, const float* __restrict__ b1i,
           const float* __restrict__ W2i, const float* __restrict__ b2i,
           const float* __restrict__ W1t, const float* __restrict__ b1t,
           const float* __restrict__ W2t, const float* __restrict__ b2t,
           float* __restrict__ small) {
    int t = threadIdx.x;
    if (t < 128) {
        float s = 0.f;
        for (int j = 0; j < 128; ++j) s += W1i[t * 128 + j] * W2i[j];
        small[t] = s;
    } else {
        int i = t - 128;
        float s = 0.f;
        for (int j = 0; j < 128; ++j) s += W1t[i * 128 + j] * W2t[j];
        small[128 + i] = s;
    }
    if (t == 0) {
        float s = 0.f;
        for (int j = 0; j < 128; ++j) s += b1i[j] * W2i[j];
        small[256] = s + b2i[0];
    }
    if (t == 1) {
        float s = 0.f;
        for (int j = 0; j < 128; ++j) s += b1t[j] * W2t[j];
        small[257] = s + b2t[0];
    }
    if (t < 4) small[260 + t] = 0.f;
}

__global__ void __launch_bounds__(256)
k_h(const float* __restrict__ ue, const float* __restrict__ p,
    const float* __restrict__ q, const float* __restrict__ small,
    float* __restrict__ hinf, float* __restrict__ hint,
    float* __restrict__ stats, int NU) {
    __shared__ float red[8][4];
    int wave = threadIdx.x >> 5, lane = threadIdx.x & 31;
    int u = blockIdx.x * 8 + wave;
    float s0 = 0.f, s1 = 0.f, s2 = 0.f, s3 = 0.f;
    if (u < NU) {
        int k = lane * 2;
        float2 x   = *(const float2*)(ue + u * DIM + k);
        float2 pv  = *(const float2*)(p  + u * DIM + k);
        float2 qv  = *(const float2*)(q  + u * DIM + k);
        float2 vi0 = *(const float2*)(small + k);
        float2 vi1 = *(const float2*)(small + 64 + k);
        float2 vt0 = *(const float2*)(small + 128 + k);
        float2 vt1 = *(const float2*)(small + 192 + k);
        float si = x.x * vi0.x + x.y * vi0.y + pv.x * vi1.x + pv.y * vi1.y;
        float st = x.x * vt0.x + x.y * vt0.y + qv.x * vt1.x + qv.y * vt1.y;
#pragma unroll
        for (int off = 16; off; off >>= 1) {
            si += __shfl_xor(si, off, 32);
            st += __shfl_xor(st, off, 32);
        }
        si += small[256];
        st += small[257];
        if (lane == 0) {
            hinf[u] = si; hint[u] = st;
            s0 = si; s1 = si * si; s2 = st; s3 = st * st;
        }
    }
    if (lane == 0) { red[wave][0] = s0; red[wave][1] = s1; red[wave][2] = s2; red[wave][3] = s3; }
    __syncthreads();
    if (threadIdx.x < 4) {
        float s = 0.f;
        for (int w = 0; w < 8; ++w) s += red[w][threadIdx.x];
        atomicAdd(stats + threadIdx.x, s);
    }
}

__global__ void __launch_bounds__(256)
k_fuse(const float* __restrict__ ue, const float* __restrict__ p,
       const float* __restrict__ q, const float* __restrict__ hinf,
       const float* __restrict__ hint, const float* __restrict__ stats,
       const float* __restrict__ gi, const float* __restrict__ bei,
       const float* __restrict__ gt, const float* __restrict__ bet,
       float* __restrict__ out, int NU) {
    int idx = blockIdx.x * 256 + threadIdx.x;
    if (idx >= NU * DIM) return;
    int u = idx >> 6;
    float inv  = 1.f / (float)NU;
    float mui  = stats[0] * inv;
    float vari = stats[1] * inv - mui * mui;
    float mut  = stats[2] * inv;
    float vart = stats[3] * inv - mut * mut;
    float si = lrelu001(gi[0] * (hinf[u] - mui) * rsqrtf(vari + 1e-5f) + bei[0]);
    float st = lrelu001(gt[0] * (hint[u] - mut) * rsqrtf(vart + 1e-5f) + bet[0]);
    float m  = fmaxf(si, st);
    float ei = __expf(si - m), et = __expf(st - m);
    float rz = 1.f / (ei + et);
    out[idx] = (ei * rz) * p[idx] + (et * rz) * q[idx] + ue[idx];
}

// ---------------- host driver ----------------
extern "C" void kernel_launch(void* const* d_in, const int* in_sizes, int n_in,
                              void* d_out, int out_size, void* d_ws, size_t ws_size,
                              hipStream_t stream) {
    const float* user_emb = (const float*)d_in[0];
    const float* item_emb = (const float*)d_in[1];
    const int* rate_src = (const int*)d_in[2];
    const int* rate_dst = (const int*)d_in[3];
    const int* rb_src   = (const int*)d_in[4];
    const int* rb_dst   = (const int*)d_in[5];
    const int* tr_src   = (const int*)d_in[6];
    const int* tr_dst   = (const int*)d_in[7];
    const float* rate_Wsrc = (const float*)d_in[8];
    const float* rate_bsrc = (const float*)d_in[9];
    const float* rate_Wdst = (const float*)d_in[10];
    const float* rate_bdst = (const float*)d_in[11];
    const float* rate_attn = (const float*)d_in[12];
    const float* rate_bias = (const float*)d_in[13];
    const float* rb_Wsrc = (const float*)d_in[14];
    const float* rb_bsrc = (const float*)d_in[15];
    const float* rb_Wdst = (const float*)d_in[16];
    const float* rb_bdst = (const float*)d_in[17];
    const float* rb_attn = (const float*)d_in[18];
    const float* rb_bias = (const float*)d_in[19];
    const float* tr_Wsrc = (const float*)d_in[20];
    const float* tr_bsrc = (const float*)d_in[21];
    const float* tr_Wdst = (const float*)d_in[22];
    const float* tr_bdst = (const float*)d_in[23];
    const float* tr_attn = (const float*)d_in[24];
    const float* tr_bias = (const float*)d_in[25];
    const float* inf_W1 = (const float*)d_in[26];
    const float* inf_b1 = (const float*)d_in[27];
    const float* inf_W2 = (const float*)d_in[28];
    const float* inf_b2 = (const float*)d_in[29];
    const float* inf_g  = (const float*)d_in[30];
    const float* inf_be = (const float*)d_in[31];
    const float* int_W1 = (const float*)d_in[32];
    const float* int_b1 = (const float*)d_in[33];
    const float* int_W2 = (const float*)d_in[34];
    const float* int_b2 = (const float*)d_in[35];
    const float* int_g  = (const float*)d_in[36];
    const float* int_be = (const float*)d_in[37];

    // ---- workspace carve-out ----
    char* ws = (char*)d_ws;
    size_t off = 0;
    auto carve = [&](size_t bytes) -> void* {
        void* p = ws + off;
        off += (bytes + 255) & ~(size_t)255;
        return p;
    };
    _Float16* user_h = (_Float16*)carve((size_t)NU_ * DIM * 2);
    _Float16* item_h = (_Float16*)carve((size_t)NI_ * DIM * 2);
    float* fsbuf = (float*)carve((size_t)NU_ * DIM * 4);
    float* fdbuf = (float*)carve((size_t)NU_ * DIM * 4);
    float* p_acc = (float*)carve((size_t)NU_ * DIM * 4);   // trust -> p_hair
    float* q_acc = (float*)carve((size_t)NU_ * DIM * 4);   // ratedby -> q_hair
    float* evalb = (float*)carve((size_t)E_ * 4);
    unsigned* menc = (unsigned*)carve((size_t)NU_ * 4);
    float* zbuf  = (float*)carve((size_t)NU_ * 4);
    float* hinf  = (float*)carve((size_t)NU_ * 4);
    float* hint  = (float*)carve((size_t)NU_ * 4);
    float* small = (float*)carve(1024);

    float* out_user = (float*)d_out;                      // [NU,64]
    float* out_item = (float*)d_out + (size_t)NU_ * DIM;  // [NI,64]

    const int g256_NU64 = (NU_ * DIM + 255) / 256;
    const int g256_NI64 = (NI_ * DIM + 255) / 256;
    const int gE_wave   = (E_ + 7) / 8;
    const int gE_thr    = (E_ + 255) / 256;

    // ---- f16 copies of embeddings ----
    k_f32_to_f16<<<g256_NU64, 256, 0, stream>>>(user_emb, user_h, NU_ * DIM);
    k_f32_to_f16<<<g256_NI64, 256, 0, stream>>>(item_emb, item_h, NI_ * DIM);

    // ---- generic per-relation GATv2 ----
    auto gat = [&](const _Float16* srcH, int Ns, const _Float16* dstH, int Nd,
                   const float* Wsrc, const float* bsrc,
                   const float* Wdst, const float* bdst,
                   const float* attn, const int* e_src, const int* e_dst,
                   float* acc) {
        k_gemm64_wmma<<<(Ns + 127) / 128, 256, 0, stream>>>(srcH, Wsrc, bsrc, fsbuf, Ns);
        k_gemm64_wmma<<<(Nd + 127) / 128, 256, 0, stream>>>(dstH, Wdst, bdst, fdbuf, Nd);
        k_init_mz<<<(Nd + 255) / 256, 256, 0, stream>>>(menc, zbuf, Nd);
        k_zero_f32<<<(Nd * DIM + 255) / 256, 256, 0, stream>>>(acc, Nd * DIM);
        k_edge_pass1<<<gE_wave, 256, 0, stream>>>(fsbuf, fdbuf, attn, e_src, e_dst,
                                                  evalb, menc, E_);
        k_edge_pass2<<<gE_thr, 256, 0, stream>>>(evalb, e_dst, menc, zbuf, E_);
        k_edge_pass3<<<gE_wave, 256, 0, stream>>>(fsbuf, evalb, zbuf, e_src, e_dst, acc, E_);
    };

    // rate: user -> item, result (+bias +item_emb) straight into d_out item region
    gat(user_h, NU_, item_h, NI_, rate_Wsrc, rate_bsrc, rate_Wdst, rate_bdst,
        rate_attn, rate_src, rate_dst, out_item);
    k_add_bias_emb<<<g256_NI64, 256, 0, stream>>>(out_item, rate_bias, item_emb, NI_ * DIM);

    // ratedby: item -> user, q_hair
    gat(item_h, NI_, user_h, NU_, rb_Wsrc, rb_bsrc, rb_Wdst, rb_bdst,
        rb_attn, rb_src, rb_dst, q_acc);
    k_add_bias_emb<<<g256_NU64, 256, 0, stream>>>(q_acc, rb_bias, nullptr, NU_ * DIM);

    // trust: user -> user, p_hair
    gat(user_h, NU_, user_h, NU_, tr_Wsrc, tr_bsrc, tr_Wdst, tr_bdst,
        tr_attn, tr_src, tr_dst, p_acc);
    k_add_bias_emb<<<g256_NU64, 256, 0, stream>>>(p_acc, tr_bias, nullptr, NU_ * DIM);

    // ---- fused attention MLP: collapse (x@W1+b1)@W2+b2 to x.v + c ----
    k_mlp_prep<<<1, 256, 0, stream>>>(inf_W1, inf_b1, inf_W2, inf_b2,
                                      int_W1, int_b1, int_W2, int_b2, small);
    k_h<<<(NU_ + 7) / 8, 256, 0, stream>>>(user_emb, p_acc, q_acc, small,
                                           hinf, hint, small + 260, NU_);
    k_fuse<<<g256_NU64, 256, 0, stream>>>(user_emb, p_acc, q_acc, hinf, hint,
                                          small + 260, inf_g, inf_be, int_g, int_be,
                                          out_user, NU_);
    (void)in_sizes; (void)n_in; (void)out_size; (void)ws_size;
}
